// GAT_30769145708915
// MI455X (gfx1250) — compile-verified, matrix operand order checked by reading, not run on hardware
//
#include <hip/hip_runtime.h>
#include <hip/hip_bf16.h>

typedef __bf16 bf16;
typedef __attribute__((ext_vector_type(8)))  __bf16 v8bf;
typedef __attribute__((ext_vector_type(16))) __bf16 v16bf;
typedef __attribute__((ext_vector_type(8)))  float  v8f;

#define BB 2
#define NNODE 2048
#define DD 128
#define HH 4
#define FF 64
#define HFD 256
#define FC1D 1024
#define FC2D 256
#define NEG_SLOPE 0.2f
#define LN_EPS 1e-5f

// ---------------------------------------------------------------------------
// WMMA fragment helpers (CDNA5 16x16x32 bf16 layouts, wave32)
// A (16x32, row-major rows): lane l (row = l&15), element e <-> k = e%8 + (e/8)*16 + (l>>4)*8
// B (32x16, via column-major Bt): lane l (col = l&15), same k pattern
// C/D f32: element r, lane l -> row = r + (l>>4)*8, col = l&15
// ---------------------------------------------------------------------------
__device__ __forceinline__ v16bf make_frag(const bf16* __restrict__ p0,
                                           const bf16* __restrict__ p1) {
  v8bf c0 = *(const v8bf*)p0;   // k = base .. base+7
  v8bf c1 = *(const v8bf*)p1;   // k = base+16 .. base+23
  v16bf f;
#pragma unroll
  for (int i = 0; i < 8; ++i) { f[i] = c0[i]; f[i + 8] = c1[i]; }
  return f;
}

// ---------------------------------------------------------------------------
// GEMM 1: H = X(bf16,[M,K]) @ Wt(bf16 col-major,[256,K]) -> h f32 [M,256]
// plus transposed bf16 copy hT[b,h,f,n] for the att@h GEMM's B operand.
// Workgroup = 8 waves; wave tile = 32 rows x 64 cols -> 8 WMMAs per 12 loads.
// ---------------------------------------------------------------------------
__global__ __launch_bounds__(256)
void gemm_h_kernel(const bf16* __restrict__ A, int lda,
                   const bf16* __restrict__ Bt, int ldb,
                   float* __restrict__ C, bf16* __restrict__ Ct, int K) {
  const int tid  = threadIdx.x;
  const int wave = tid >> 5, lane = tid & 31;
  const int half = lane >> 4, l16 = lane & 15;
  const int rowBase = blockIdx.x * 256 + wave * 32;
  const int colBase = blockIdx.y * 64;

  v8f acc[2][4] = {};
  const bf16* Arow0 = A + (size_t)(rowBase + l16) * lda + half * 8;
  const bf16* Arow1 = Arow0 + (size_t)16 * lda;
  for (int kt = 0; kt < K; kt += 32) {
    __builtin_prefetch(Arow0 + kt + 128, 0, 1);
    __builtin_prefetch(Arow1 + kt + 128, 0, 1);
    v16bf a0 = make_frag(Arow0 + kt, Arow0 + kt + 16);
    v16bf a1 = make_frag(Arow1 + kt, Arow1 + kt + 16);
#pragma unroll
    for (int cb = 0; cb < 4; ++cb) {
      const bf16* bp = Bt + (size_t)(colBase + cb * 16 + l16) * ldb + kt + half * 8;
      v16bf bfrag = make_frag(bp, bp + 16);
      acc[0][cb] = __builtin_amdgcn_wmma_f32_16x16x32_bf16(
          false, a0, false, bfrag, (short)0, acc[0][cb], false, false);
      acc[1][cb] = __builtin_amdgcn_wmma_f32_16x16x32_bf16(
          false, a1, false, bfrag, (short)0, acc[1][cb], false, false);
    }
  }
#pragma unroll
  for (int rb = 0; rb < 2; ++rb) {
#pragma unroll
    for (int cb = 0; cb < 4; ++cb) {
      int col = colBase + cb * 16 + l16;
      int hh = col >> 6, f = col & 63;
#pragma unroll
      for (int r = 0; r < 8; ++r) {
        int row = rowBase + rb * 16 + half * 8 + r;
        float v = acc[rb][cb][r];
        C[(size_t)row * HFD + col] = v;
        int b = row >> 11, n = row & (NNODE - 1);
        Ct[(((size_t)(b * HH + hh)) * FF + f) * NNODE + n] = (bf16)v;
      }
    }
  }
}

// ---------------------------------------------------------------------------
// GEMM 2 (batched over z = b*H + h): X = relu(att(bf16,[N,N]) @ h) -> [B,N,HF]
// B operand comes from hT (col-major per head, L2-resident), so all K loads
// are contiguous.  att (A operand) streams -> prefetch ahead.
// ---------------------------------------------------------------------------
__global__ __launch_bounds__(256)
void gemm_att_kernel(const bf16* __restrict__ Att, const bf16* __restrict__ Ht,
                     float* __restrict__ Xf, bf16* __restrict__ Xb) {
  const int z = blockIdx.z;
  const int b = z >> 2, hh = z & 3;
  const bf16* A  = Att + (size_t)z * NNODE * NNODE;
  const bf16* Bt = Ht  + (size_t)z * FF * NNODE;

  const int tid  = threadIdx.x;
  const int wave = tid >> 5, lane = tid & 31;
  const int half = lane >> 4, l16 = lane & 15;
  const int rowBase = blockIdx.x * 256 + wave * 32;

  v8f acc[2][4] = {};
  const bf16* Arow0 = A + (size_t)(rowBase + l16) * NNODE + half * 8;
  const bf16* Arow1 = Arow0 + (size_t)16 * NNODE;
  for (int kt = 0; kt < NNODE; kt += 32) {
    __builtin_prefetch(Arow0 + kt + 128, 0, 1);
    __builtin_prefetch(Arow1 + kt + 128, 0, 1);
    v16bf a0 = make_frag(Arow0 + kt, Arow0 + kt + 16);
    v16bf a1 = make_frag(Arow1 + kt, Arow1 + kt + 16);
#pragma unroll
    for (int cb = 0; cb < 4; ++cb) {
      const bf16* bp = Bt + (size_t)(cb * 16 + l16) * NNODE + kt + half * 8;
      v16bf bfrag = make_frag(bp, bp + 16);
      acc[0][cb] = __builtin_amdgcn_wmma_f32_16x16x32_bf16(
          false, a0, false, bfrag, (short)0, acc[0][cb], false, false);
      acc[1][cb] = __builtin_amdgcn_wmma_f32_16x16x32_bf16(
          false, a1, false, bfrag, (short)0, acc[1][cb], false, false);
    }
  }
#pragma unroll
  for (int rb = 0; rb < 2; ++rb) {
#pragma unroll
    for (int cb = 0; cb < 4; ++cb) {
      int col = cb * 16 + l16;  // 0..63 within this head
#pragma unroll
      for (int r = 0; r < 8; ++r) {
        int row = rowBase + rb * 16 + half * 8 + r;
        float v = fmaxf(acc[rb][cb][r], 0.0f);
        size_t oi = ((size_t)b * NNODE + row) * HFD + hh * FF + col;
        Xf[oi] = v;
        if (Xb) Xb[oi] = (bf16)v;
      }
    }
  }
}

// ---------------------------------------------------------------------------
// Small kernels
// ---------------------------------------------------------------------------
__global__ void conv_bf16_kernel(const float* __restrict__ in, bf16* __restrict__ out, int n) {
  int i = blockIdx.x * blockDim.x + threadIdx.x;
  if (i < n) out[i] = (bf16)in[i];
}

// W1 [H,D,F] f32 -> col-major bf16 [256][D]
__global__ void conv_w1t_kernel(const float* __restrict__ W, bf16* __restrict__ Wt) {
  int i = blockIdx.x * blockDim.x + threadIdx.x;
  if (i >= HFD * DD) return;
  int col = i / DD, k = i % DD;
  int hh = col >> 6, f = col & 63;
  Wt[i] = (bf16)W[((size_t)hh * DD + k) * FF + f];
}

// W2 [H,HF,F] f32 -> col-major bf16 [256][256]
__global__ void conv_w2t_kernel(const float* __restrict__ W, bf16* __restrict__ Wt) {
  int i = blockIdx.x * blockDim.x + threadIdx.x;
  if (i >= HFD * HFD) return;
  int col = i / HFD, k = i % HFD;
  int hh = col >> 6, f = col & 63;
  Wt[i] = (bf16)W[((size_t)hh * HFD + k) * FF + f];
}

__global__ void x0_kernel(const float* __restrict__ x, float* __restrict__ gat) {
  int idx = blockIdx.x * blockDim.x + threadIdx.x;
  if (idx >= BB * NNODE) return;
  int b = idx / NNODE, n = idx % NNODE;
  const float* xr = x + (size_t)idx * DD;
  float s = 0.f;
  for (int k = 0; k < DD; ++k) s += xr[k];
  gat[(size_t)b * (3 * NNODE) + n] = s * (1.0f / DD);
}

// es/ed dots: es[(b*H+h)*N+n] = h_row . a_src
__global__ void esed_kernel(const float* __restrict__ h, const float* __restrict__ a_src,
                            const float* __restrict__ a_dst, float* __restrict__ es,
                            float* __restrict__ ed) {
  int idx = blockIdx.x * blockDim.x + threadIdx.x;
  if (idx >= BB * HH * NNODE) return;
  int b = idx / (HH * NNODE);
  int r = idx % (HH * NNODE);
  int hh = r / NNODE, n = r % NNODE;
  const float* hp = h + ((size_t)b * NNODE + n) * HFD + hh * FF;
  const float* as = a_src + hh * FF;
  const float* ad = a_dst + hh * FF;
  float s = 0.f, d = 0.f;
  for (int f = 0; f < FF; ++f) { float hv = hp[f]; s += hv * as[f]; d += hv * ad[f]; }
  es[idx] = s; ed[idx] = d;
}

// Fused masked-leaky-relu + softmax per row (b,i), looping 4 heads.
// Writes bf16 att rows and accumulates 0.125*sum_h att into attention_map.
__global__ __launch_bounds__(256)
void att_kernel(const float* __restrict__ es, const float* __restrict__ ed,
                const int* __restrict__ adj, bf16* __restrict__ attb,
                float* __restrict__ amap, int addMode) {
  __shared__ float red[256];
  const int i = blockIdx.x, b = blockIdx.y, tid = threadIdx.x;
  float acc[8] = {};
  for (int hh = 0; hh < HH; ++hh) {
    int z = b * HH + hh;
    float esv = es[(size_t)z * NNODE + i];
    float ev[8];
    float mx = -3.0e38f;
#pragma unroll
    for (int c = 0; c < 8; ++c) {
      int j = c * 256 + tid;
      float e = esv + ed[(size_t)z * NNODE + j];
      e = e > 0.f ? e : NEG_SLOPE * e;
      if (adj[(size_t)i * NNODE + j] == 0) e = -9.0e15f;
      ev[c] = e;
      mx = fmaxf(mx, e);
    }
    red[tid] = mx; __syncthreads();
    for (int s = 128; s > 0; s >>= 1) { if (tid < s) red[tid] = fmaxf(red[tid], red[tid + s]); __syncthreads(); }
    float M = red[0]; __syncthreads();
    float sum = 0.f;
#pragma unroll
    for (int c = 0; c < 8; ++c) { float p = __expf(ev[c] - M); ev[c] = p; sum += p; }
    red[tid] = sum; __syncthreads();
    for (int s = 128; s > 0; s >>= 1) { if (tid < s) red[tid] += red[tid + s]; __syncthreads(); }
    float inv = 1.0f / red[0]; __syncthreads();
    bf16* ao = attb + ((size_t)z * NNODE + i) * NNODE;
#pragma unroll
    for (int c = 0; c < 8; ++c) {
      int j = c * 256 + tid;
      float a = ev[c] * inv;
      ao[j] = (bf16)a;
      acc[c] += a;
    }
  }
  float* mo = amap + ((size_t)b * NNODE + i) * NNODE;
#pragma unroll
  for (int c = 0; c < 8; ++c) {
    int j = c * 256 + tid;
    float v = acc[c] * 0.125f;  // mean over H=4 heads, then *0.5
    if (addMode) mo[j] += v; else mo[j] = v;
  }
}

__global__ void pool_kernel(const float* __restrict__ x, const float* __restrict__ w,
                            const float* __restrict__ bsc, float* __restrict__ gat, int section) {
  int idx = blockIdx.x * blockDim.x + threadIdx.x;
  if (idx >= BB * NNODE) return;
  int b = idx / NNODE, n = idx % NNODE;
  const float* xr = x + (size_t)idx * HFD;
  float s = bsc[0];
  for (int k = 0; k < HFD; ++k) s += xr[k] * w[k];
  gat[(size_t)b * (3 * NNODE) + section * NNODE + n] = s;
}

__global__ void fc_kernel(const float* __restrict__ in, const float* __restrict__ W,
                          const float* __restrict__ bias, float* __restrict__ out,
                          int Kdim, int Odim) {
  int idx = blockIdx.x * blockDim.x + threadIdx.x;
  if (idx >= BB * Odim) return;
  int b = idx / Odim, o = idx % Odim;
  const float* ir = in + (size_t)b * Kdim;
  float s = bias[o];
  for (int k = 0; k < Kdim; ++k) s += ir[k] * W[(size_t)k * Odim + o];
  out[idx] = s;
}

__global__ __launch_bounds__(256)
void ln_relu_kernel(const float* __restrict__ in, const float* __restrict__ g,
                    const float* __restrict__ beta, float* __restrict__ out, int C) {
  __shared__ float red[256];
  const int b = blockIdx.x, tid = threadIdx.x;
  const float* ir = in + (size_t)b * C;
  float s = 0.f, sq = 0.f;
  for (int c = tid; c < C; c += 256) { float v = ir[c]; s += v; sq += v * v; }
  red[tid] = s; __syncthreads();
  for (int st = 128; st > 0; st >>= 1) { if (tid < st) red[tid] += red[tid + st]; __syncthreads(); }
  float mean = red[0] / C; __syncthreads();
  red[tid] = sq; __syncthreads();
  for (int st = 128; st > 0; st >>= 1) { if (tid < st) red[tid] += red[tid + st]; __syncthreads(); }
  float var = red[0] / C - mean * mean; __syncthreads();
  float inv = rsqrtf(var + LN_EPS);
  for (int c = tid; c < C; c += 256) {
    float v = (ir[c] - mean) * inv * g[c] + beta[c];
    out[(size_t)b * C + c] = fmaxf(v, 0.f);
  }
}

__global__ void cls_kernel(const float* __restrict__ fc, const float* __restrict__ W,
                           const float* __restrict__ bias, float* __restrict__ out) {
  int idx = threadIdx.x;
  if (idx >= BB * 2) return;
  int b = idx >> 1, l = idx & 1;
  float s = bias[l];
  for (int k = 0; k < FC2D; ++k) s += fc[b * FC2D + k] * W[k * 2 + l];
  out[idx] = s;
}

// ---------------------------------------------------------------------------
extern "C" void kernel_launch(void* const* d_in, const int* in_sizes, int n_in,
                              void* d_out, int out_size, void* d_ws, size_t ws_size,
                              hipStream_t stream) {
  const float* x    = (const float*)d_in[0];
  const int*   adj  = (const int*)d_in[1];
  const float* W1   = (const float*)d_in[3];
  const float* a1s  = (const float*)d_in[4];
  const float* a1d  = (const float*)d_in[5];
  const float* p1w  = (const float*)d_in[6];
  const float* p1b  = (const float*)d_in[7];
  const float* W2   = (const float*)d_in[8];
  const float* a2s  = (const float*)d_in[9];
  const float* a2d  = (const float*)d_in[10];
  const float* p2w  = (const float*)d_in[11];
  const float* p2b  = (const float*)d_in[12];
  const float* fc1W = (const float*)d_in[13];
  const float* fc1b = (const float*)d_in[14];
  const float* ln1g = (const float*)d_in[15];
  const float* ln1b = (const float*)d_in[16];
  const float* fc2W = (const float*)d_in[17];
  const float* fc2b = (const float*)d_in[18];
  const float* ln2g = (const float*)d_in[19];
  const float* ln2b = (const float*)d_in[20];
  const float* clsW = (const float*)d_in[21];
  const float* clsb = (const float*)d_in[22];

  float* out    = (float*)d_out;
  float* gat    = out;                         // [B, 3N]
  float* fcf    = out + BB * 3 * NNODE;        // [B, FC2]
  float* logits = fcf + BB * FC2D;             // [B, 2]
  float* amap   = logits + BB * 2;             // [B, N, N]

  char* wptr = (char*)d_ws;
  auto carve = [&](size_t bytes) -> void* {
    void* p = (void*)wptr;
    wptr += (bytes + 255) & ~(size_t)255;
    return p;
  };
  bf16*  xb   = (bf16*)carve(sizeof(bf16) * BB * NNODE * DD);
  bf16*  w1t  = (bf16*)carve(sizeof(bf16) * HFD * DD);
  bf16*  w2t  = (bf16*)carve(sizeof(bf16) * HFD * HFD);
  float* hft  = (float*)carve(sizeof(float) * BB * NNODE * HFD);
  bf16*  hTb  = (bf16*)carve(sizeof(bf16) * BB * HH * FF * NNODE);
  float* es   = (float*)carve(sizeof(float) * BB * HH * NNODE);
  float* ed   = (float*)carve(sizeof(float) * BB * HH * NNODE);
  bf16*  attb = (bf16*)carve(sizeof(bf16) * (size_t)BB * HH * NNODE * NNODE);  // reused both layers
  float* x1f  = (float*)carve(sizeof(float) * BB * NNODE * HFD);
  bf16*  x1b  = (bf16*)carve(sizeof(bf16) * BB * NNODE * HFD);
  float* x2f  = (float*)carve(sizeof(float) * BB * NNODE * HFD);
  float* t1   = (float*)carve(sizeof(float) * BB * FC1D);
  float* y1   = (float*)carve(sizeof(float) * BB * FC1D);
  float* t2   = (float*)carve(sizeof(float) * BB * FC2D);

  // -- precision staging + x0 --
  {
    int n = BB * NNODE * DD;
    conv_bf16_kernel<<<(n + 255) / 256, 256, 0, stream>>>(x, xb, n);
  }
  conv_w1t_kernel<<<(HFD * DD + 255) / 256, 256, 0, stream>>>(W1, w1t);
  conv_w2t_kernel<<<(HFD * HFD + 255) / 256, 256, 0, stream>>>(W2, w2t);
  x0_kernel<<<(BB * NNODE + 255) / 256, 256, 0, stream>>>(x, gat);

  // -- GAT layer 1 --
  gemm_h_kernel<<<dim3(BB * NNODE / 256, HFD / 64), 256, 0, stream>>>(xb, DD, w1t, DD, hft, hTb, DD);
  esed_kernel<<<(BB * HH * NNODE + 255) / 256, 256, 0, stream>>>(hft, a1s, a1d, es, ed);
  att_kernel<<<dim3(NNODE, BB), 256, 0, stream>>>(es, ed, adj, attb, amap, 0);
  gemm_att_kernel<<<dim3(NNODE / 256, 1, BB * HH), 256, 0, stream>>>(attb, hTb, x1f, x1b);
  pool_kernel<<<(BB * NNODE + 255) / 256, 256, 0, stream>>>(x1f, p1w, p1b, gat, 1);

  // -- GAT layer 2 (reuses hft/hTb/es/ed/attb) --
  gemm_h_kernel<<<dim3(BB * NNODE / 256, HFD / 64), 256, 0, stream>>>(x1b, HFD, w2t, HFD, hft, hTb, HFD);
  esed_kernel<<<(BB * HH * NNODE + 255) / 256, 256, 0, stream>>>(hft, a2s, a2d, es, ed);
  att_kernel<<<dim3(NNODE, BB), 256, 0, stream>>>(es, ed, adj, attb, amap, 1);
  gemm_att_kernel<<<dim3(NNODE / 256, 1, BB * HH), 256, 0, stream>>>(attb, hTb, x2f, (bf16*)nullptr);
  pool_kernel<<<(BB * NNODE + 255) / 256, 256, 0, stream>>>(x2f, p2w, p2b, gat, 2);

  // -- MLP head (B=2; VALU is the right tool here) --
  fc_kernel<<<(BB * FC1D + 255) / 256, 256, 0, stream>>>(gat, fc1W, fc1b, t1, 3 * NNODE, FC1D);
  ln_relu_kernel<<<BB, 256, 0, stream>>>(t1, ln1g, ln1b, y1, FC1D);
  fc_kernel<<<(BB * FC2D + 255) / 256, 256, 0, stream>>>(y1, fc2W, fc2b, t2, FC1D, FC2D);
  ln_relu_kernel<<<BB, 256, 0, stream>>>(t2, ln2g, ln2b, fcf, FC2D);
  cls_kernel<<<1, 64, 0, stream>>>(fcf, clsW, clsb, logits);
}
